// CompILE_49125835931949
// MI455X (gfx1250) — matrix operand order repaired and checked by reading, not compiled, not run on hardware
//
#include <hip/hip_runtime.h>
#include <math.h>

// ---------------- problem dims / constants ----------------
constexpr int NB = 256, NT = 256, NV = 128, NH = 256, NA = 8, NE = 16, NS = 4;
constexpr int NBT = NB * NT;
constexpr int NG  = 4 * NH;              // 1024 gate width
constexpr float C_NEG_INF = -1e30f;
constexpr float C_EPS     = 1e-17f;
constexpr float C_LN_EPS  = 1e-5f;
constexpr float C_BETA_B  = 0.1f;
constexpr float C_BETA_Z  = 0.1f;
constexpr float C_PRIOR   = 3.0f;

typedef __bf16 bf16;
typedef __attribute__((ext_vector_type(16))) __bf16       v16bf;
typedef __attribute__((ext_vector_type(8)))  float        v8f;
typedef __attribute__((ext_vector_type(4)))  unsigned int u32x4;
typedef __attribute__((ext_vector_type(8)))  int          i32x8;
typedef __attribute__((ext_vector_type(4)))  int          i32x4;

// ---- Tensor Data Mover availability (device pass only; arity differs per toolchain) ----
#if defined(__gfx1250__) && __has_builtin(__builtin_amdgcn_tensor_load_to_lds) && \
    __has_builtin(__builtin_amdgcn_s_wait_tensorcnt)
  #define USE_TDM 1
  #if __has_include(<hip/amd_detail/amd_gfx1250_TDM.h>)
    #define TDM_SIX_ARG 1
  #else
    #define TDM_SIX_ARG 0
  #endif
#else
  #define USE_TDM 0
#endif

union Frag { v16bf v; struct { u32x4 lo, hi; } q; };

__device__ __forceinline__ v8f v8f_zero() {
  v8f v = {0.f,0.f,0.f,0.f,0.f,0.f,0.f,0.f}; return v;
}

__device__ __forceinline__ v8f wmma_bf(v16bf a, v16bf b, v8f c) {
  return __builtin_amdgcn_wmma_f32_16x16x32_bf16(false, a, false, b, (short)0, c, false, false);
}

// A fragment (16x32 bf16, row-major source): row m = lane&15.
// lanes 0-15: elems 0..7 = K[kb..kb+7], elems 8..15 = K[kb+16..kb+23]
// lanes 16-31: same shifted by 8 in K.
__device__ __forceinline__ v16bf load_a_bf(const bf16* base, int lda, int kb) {
  int lane = threadIdx.x & 31, m = lane & 15, hi = lane >> 4;
  const bf16* p = base + (size_t)m * lda + kb + hi * 8;
  Frag f;
  f.q.lo = *reinterpret_cast<const u32x4*>(p);
  f.q.hi = *reinterpret_cast<const u32x4*>(p + 16);
  return f.v;
}

__device__ __forceinline__ v16bf load_a_f32(const float* base, int lda, int kb) {
  int lane = threadIdx.x & 31, m = lane & 15, hi = lane >> 4;
  const float* p = base + (size_t)m * lda + kb + hi * 8;
  v16bf v;
#pragma unroll
  for (int i = 0; i < 8; ++i) { v[i] = (bf16)p[i]; v[8 + i] = (bf16)p[16 + i]; }
  return v;
}

// B fragment (32x16 bf16) taken from row-major W[N][K] (computing A @ W^T):
// col n = lane&15; lanes 0-15: elems 0..7 = K[kb..kb+7], elems 8..15 = K[kb+16..kb+23]; lanes 16-31 shifted by 8.
__device__ __forceinline__ v16bf load_b_bf(const bf16* W, int ldw, int ncol, int kb) {
  int lane = threadIdx.x & 31, n = lane & 15, hi = lane >> 4;
  const bf16* p = W + (size_t)(ncol + n) * ldw + kb + hi * 8;
  Frag f;
  f.q.lo = *reinterpret_cast<const u32x4*>(p);
  f.q.hi = *reinterpret_cast<const u32x4*>(p + 16);
  return f.v;
}

#if USE_TDM
// One TDM descriptor move: 2D tile of 16 rows x 1024 bf16 (2KB/row),
// global row stride NT*NG elements, packed contiguously into LDS at lds_off.
// D# layout per CDNA5 ISA ch.8: group0 {count=1, lds_addr, global_addr, type=2},
// group1 {data_size=2B, tensor dims, tile dims, dim0 stride}; groups 2/3 unused (2D).
__device__ __forceinline__ void tdm_load_xtile(unsigned lds_off, const bf16* gptr) {
  unsigned long long ga = (unsigned long long)(size_t)gptr;
  u32x4 g0 = { 1u,                                   // count=1, user mode, no gather
               lds_off,                              // lds_addr (bytes)
               (unsigned)(ga & 0xFFFFFFFFu),         // global_addr[31:0]
               (unsigned)((ga >> 32) & 0x01FFFFFFu)  // global_addr[56:32]
                 | (2u << 30) };                     // type = 2 ("image")
  i32x8 g1;
  g1[0] = (int)(1u << 16);            // workgroup_mask=0 (not in cluster), data_size=1 (2B)
  g1[1] = 0;                          // no atomic barrier; tensor_dim0[15:0]=0
  g1[2] = (int)((16u << 16) | 0x10u); // tensor_dim0=1<<20 (no clip), tensor_dim1=16
  g1[3] = (int)(1024u << 16);         // tile_dim0 = 1024 elements (2KB row)
  g1[4] = 16;                         // tile_dim1 = 16 rows, tile_dim2 unused
  g1[5] = NT * NG;                    // tensor_dim0_stride[31:0] = 262144 elements
  g1[6] = 0;                          // stride[47:32]=0, dim1_stride=0 (2D)
  g1[7] = 0;
  i32x4 z4 = {0, 0, 0, 0};
#if TDM_SIX_ARG
  i32x8 z8 = {0, 0, 0, 0, 0, 0, 0, 0};
  __builtin_amdgcn_tensor_load_to_lds(g0, g1, z4, z4, z8, 0);
#else
  __builtin_amdgcn_tensor_load_to_lds(g0, g1, z4, z4, 0);
#endif
}
#endif

__device__ __forceinline__ float block_max256(float v, float* red) {
  int t = threadIdx.x;
  red[t] = v; __syncthreads();
  for (int o = 128; o > 0; o >>= 1) { if (t < o) red[t] = fmaxf(red[t], red[t + o]); __syncthreads(); }
  float r = red[0]; __syncthreads(); return r;
}
__device__ __forceinline__ float block_sum256(float v, float* red) {
  int t = threadIdx.x;
  red[t] = v; __syncthreads();
  for (int o = 128; o > 0; o >>= 1) { if (t < o) red[t] += red[t + o]; __syncthreads(); }
  float r = red[0]; __syncthreads(); return r;
}

// ---------------- small utility kernels ----------------
__global__ void k_f2bf(const float* __restrict__ src, bf16* __restrict__ dst, int n) {
  int i = blockIdx.x * blockDim.x + threadIdx.x;
  if (i < n) dst[i] = (bf16)src[i];
}

__global__ void k_init(float* __restrict__ maskws, float* __restrict__ logcum,
                       float* __restrict__ accs) {
  int i = blockIdx.x * blockDim.x + threadIdx.x;
  if (i < NBT) { maskws[i] = 1.f; logcum[i] = 0.f; }
  if (i == 0) { accs[0] = 0.f; accs[1] = 0.f; accs[2] = 0.f; accs[3] = 0.f; }
}

// rank-1 precomputes: env/act contributions through W_enc and W_d1 + Poisson log-prior
__global__ void k_prep(const float* __restrict__ emb_env, const float* __restrict__ emb_act,
                       const float* __restrict__ W_enc, const float* __restrict__ W_d1,
                       float* __restrict__ E2, float* __restrict__ A3,
                       float* __restrict__ Denv, float* __restrict__ logprior) {
  __shared__ float red[256];
  int n = threadIdx.x;
  for (int e = 0; e < NE; ++e) {
    float s = 0.f;
    for (int k = 0; k < NH; ++k) s += emb_env[e * NH + k] * W_enc[n * (3 * NH) + NH + k];
    E2[e * NH + n] = s;
  }
  for (int a = 0; a < NA; ++a) {
    float s = 0.f;
    for (int k = 0; k < NH; ++k) s += emb_act[a * NH + k] * W_enc[n * (3 * NH) + 2 * NH + k];
    A3[a * NH + n] = s;
  }
  for (int e = 0; e < NE; ++e) {
    float s = 0.f;
    for (int k = 0; k < NH; ++k) s += emb_env[e * NH + k] * W_d1[n * (3 * NH) + NH + k];
    Denv[e * NH + n] = s;
  }
  float v = logf(C_PRIOR) * (float)(n + 1) - C_PRIOR - lgammaf((float)(n + 2));
  float m = block_max256(v, red);
  float den = block_sum256(__expf(v - m), red);
  logprior[n] = v - (m + logf(den));
}

// ---------------- GEMM kernels (bf16 WMMA) ----------------
// obs_emb = features @ W_obs^T + b_obs -> bf16 [BT,H]
__global__ void k_obs(const float* __restrict__ feat, const bf16* __restrict__ Wobs,
                      const float* __restrict__ b_obs, bf16* __restrict__ obs_bf) {
  int m0 = blockIdx.x * 16;
  int w  = threadIdx.x >> 5;
  int nc0 = (w * 2) * 16, nc1 = nc0 + 16;
  v8f acc0 = v8f_zero(), acc1 = v8f_zero();
  const float* A = feat + (size_t)m0 * NV;
  for (int kk = 0; kk < NV; kk += 32) {
    v16bf a = load_a_f32(A, NV, kk);
    acc0 = wmma_bf(a, load_b_bf(Wobs, NV, nc0, kk), acc0);
    acc1 = wmma_bf(a, load_b_bf(Wobs, NV, nc1, kk), acc1);
  }
  int col = threadIdx.x & 15, rb = ((threadIdx.x & 31) >> 4) * 8;
#pragma unroll
  for (int r = 0; r < 8; ++r) {
    int row = rb + r;
    obs_bf[(size_t)(m0 + row) * NH + nc0 + col] = (bf16)(acc0[r] + b_obs[nc0 + col]);
    obs_bf[(size_t)(m0 + row) * NH + nc1 + col] = (bf16)(acc1[r] + b_obs[nc1 + col]);
  }
}

// embeddings = LayerNorm(relu(obs@W1^T + E2[env] + A3[act] + b_enc)) -> bf16 [BT,H]
__global__ void k_enc(const bf16* __restrict__ obs_bf, const bf16* __restrict__ Wenc,
                      const float* __restrict__ b_enc, const float* __restrict__ E2,
                      const float* __restrict__ A3, const int* __restrict__ action,
                      const int* __restrict__ env_id, bf16* __restrict__ emb_bf) {
  __shared__ float tile[16][260];
  __shared__ float mu[16], rs[16];
  int m0 = blockIdx.x * 16;
  int w  = threadIdx.x >> 5;
  int nc0 = (w * 2) * 16, nc1 = nc0 + 16;
  v8f acc0 = v8f_zero(), acc1 = v8f_zero();
  const bf16* A = obs_bf + (size_t)m0 * NH;
  for (int kk = 0; kk < NH; kk += 32) {
    v16bf a = load_a_bf(A, NH, kk);
    acc0 = wmma_bf(a, load_b_bf(Wenc, 3 * NH, nc0, kk), acc0);
    acc1 = wmma_bf(a, load_b_bf(Wenc, 3 * NH, nc1, kk), acc1);
  }
  int col = threadIdx.x & 15, rb = ((threadIdx.x & 31) >> 4) * 8;
#pragma unroll
  for (int r = 0; r < 8; ++r) {
    int row = rb + r, bt = m0 + row, b = bt >> 8;
    int e = env_id[b], ac = action[bt];
    float v0 = acc0[r] + b_enc[nc0 + col] + E2[e * NH + nc0 + col] + A3[ac * NH + nc0 + col];
    float v1 = acc1[r] + b_enc[nc1 + col] + E2[e * NH + nc1 + col] + A3[ac * NH + nc1 + col];
    tile[row][nc0 + col] = fmaxf(v0, 0.f);
    tile[row][nc1 + col] = fmaxf(v1, 0.f);
  }
  __syncthreads();
  if (threadIdx.x < 16) {
    int row = threadIdx.x;
    float s = 0.f, ss = 0.f;
    for (int i = 0; i < NH; ++i) { float x = tile[row][i]; s += x; ss += x * x; }
    float m = s / NH, var = ss / NH - m * m;
    mu[row] = m; rs[row] = rsqrtf(var + C_LN_EPS);
  }
  __syncthreads();
  int row = threadIdx.x >> 4, c0 = (threadIdx.x & 15) * 16;
  for (int c = 0; c < 16; ++c)
    emb_bf[(size_t)(m0 + row) * NH + c0 + c] = (bf16)((tile[row][c0 + c] - mu[row]) * rs[row]);
}

// xgates = embeddings @ W_ih^T + b_ih + b_hh -> bf16 [BT, 4H]  (shared by all 4 segments)
__global__ void k_xproj(const bf16* __restrict__ emb_bf, const bf16* __restrict__ Wih,
                        const float* __restrict__ b_ih, const float* __restrict__ b_hh,
                        bf16* __restrict__ xg) {
  int m0 = blockIdx.x * 16;
  int n0 = blockIdx.y * 256;
  int w  = threadIdx.x >> 5;
  int nc0 = n0 + (w * 2) * 16, nc1 = nc0 + 16;
  v8f acc0 = v8f_zero(), acc1 = v8f_zero();
  const bf16* A = emb_bf + (size_t)m0 * NH;
  for (int kk = 0; kk < NH; kk += 32) {
    v16bf a = load_a_bf(A, NH, kk);
    acc0 = wmma_bf(a, load_b_bf(Wih, NH, nc0, kk), acc0);
    acc1 = wmma_bf(a, load_b_bf(Wih, NH, nc1, kk), acc1);
  }
  int col = threadIdx.x & 15, rb = ((threadIdx.x & 31) >> 4) * 8;
#pragma unroll
  for (int r = 0; r < 8; ++r) {
    int row = rb + r;
    xg[(size_t)(m0 + row) * NG + nc0 + col] = (bf16)(acc0[r] + b_ih[nc0 + col] + b_hh[nc0 + col]);
    xg[(size_t)(m0 + row) * NG + nc1 + col] = (bf16)(acc1[r] + b_ih[nc1 + col] + b_hh[nc1 + col]);
  }
}

// masked LSTM over T: 16 blocks (batch tiles of 16), 512 threads = 16 waves.
// Wave j owns hidden slice [16j,16j+16) and gate tiles i/f/g/o at cols j,j+16,j+32,j+48 (x16).
// Cell state cc stays in registers; h_t round-trips through LDS (bf16).
// Next step's x-gate tile (16x2KB) is DMA'd into LDS by the Tensor Data Mover,
// overlapping TDM traffic with the 32 WMMAs + activations of the current step.
__global__ void k_lstm(const bf16* __restrict__ xg, const bf16* __restrict__ Whh,
                       const float* __restrict__ maskws, bf16* __restrict__ enc_bf) {
  __shared__ bf16 hh[16][264];        // padded stride vs 64 banks
  __shared__ bf16 xs[16][NG];         // staged gate inputs for this t
  __shared__ float mcol[16];
  int b0   = blockIdx.x * 16;
  int w    = threadIdx.x >> 5;        // h-tile / wave id
  int lane = threadIdx.x & 31;
  int col  = lane & 15, rb = (lane >> 4) * 8;
  for (int i = threadIdx.x; i < 16 * 264; i += 512) (&hh[0][0])[i] = (bf16)0.f;
  v8f cc = v8f_zero();
#if USE_TDM
  unsigned xs_lds = (unsigned)(size_t)(&xs[0][0]);
  if (w == 0)
    tdm_load_xtile(xs_lds, xg + (size_t)b0 * NT * NG);   // prefetch t = 0
#endif
  __syncthreads();
  for (int t = 0; t < NT; ++t) {
#if USE_TDM
    if (w == 0) __builtin_amdgcn_s_wait_tensorcnt(0);    // tile(t) landed in LDS
    if (threadIdx.x < 16) mcol[threadIdx.x] = maskws[(b0 + threadIdx.x) * NT + t];
    __syncthreads();                                     // publish xs(t) + mask
#else
    { // fallback: cooperative synchronous staging (coalesced b128)
      int row = threadIdx.x >> 5, tt = threadIdx.x & 31;
      const u32x4* src = reinterpret_cast<const u32x4*>(xg + ((size_t)(b0 + row) * NT + t) * NG);
      u32x4* dst = reinterpret_cast<u32x4*>(&xs[row][0]);
      for (int q = tt; q < NG * 2 / 16; q += 32) dst[q] = src[q];
      if (t + 1 < NT)
        __builtin_prefetch(xg + ((size_t)(b0 + row) * NT + t + 1) * NG, 0, 1);
      if (threadIdx.x < 16) mcol[threadIdx.x] = maskws[(b0 + threadIdx.x) * NT + t];
    }
    __syncthreads();
#endif
    v8f gi, gf, gg, go;
#pragma unroll
    for (int r = 0; r < 8; ++r) {
      int row = rb + r, h = w * 16 + col;
      gi[r] = (float)xs[row][h];
      gf[r] = (float)xs[row][NH + h];
      gg[r] = (float)xs[row][2 * NH + h];
      go[r] = (float)xs[row][3 * NH + h];
    }
#if USE_TDM
    __syncthreads();                                     // xs(t) fully consumed
    if (w == 0 && t + 1 < NT)
      tdm_load_xtile(xs_lds, xg + ((size_t)b0 * NT + (t + 1)) * NG);  // overlap with WMMA
#endif
    for (int kk = 0; kk < NH; kk += 32) {
      v16bf a = load_a_bf(&hh[0][0], 264, kk);
      gi = wmma_bf(a, load_b_bf(Whh, NH, w * 16,           kk), gi);
      gf = wmma_bf(a, load_b_bf(Whh, NH, NH + w * 16,      kk), gf);
      gg = wmma_bf(a, load_b_bf(Whh, NH, 2 * NH + w * 16,  kk), gg);
      go = wmma_bf(a, load_b_bf(Whh, NH, 3 * NH + w * 16,  kk), go);
    }
    __syncthreads();                  // all reads of hh done before updates
#pragma unroll
    for (int r = 0; r < 8; ++r) {
      int row = rb + r;
      float mk = mcol[row];
      float i_ = 1.f / (1.f + __expf(-gi[r]));
      float f_ = 1.f / (1.f + __expf(-gf[r]));
      float g_ = tanhf(gg[r]);
      float o_ = 1.f / (1.f + __expf(-go[r]));
      float c  = f_ * cc[r] + i_ * g_;
      float hv = o_ * tanhf(c);
      hv *= mk; c *= mk;
      cc[r] = c;
      int h = w * 16 + col;
      hh[row][h] = (bf16)hv;
      enc_bf[((size_t)(b0 + row) * NT + t) * NH + h] = (bf16)hv;
    }
    __syncthreads();
  }
}

// boundary logits lb = relu(enc@W_b1^T + b_b1) @ W_b2^T + b_b2   -> f32 [B,T]
__global__ void k_bound(const bf16* __restrict__ enc_bf, const bf16* __restrict__ Wb1,
                        const float* __restrict__ b_b1, const float* __restrict__ W_b2,
                        const float* __restrict__ b_b2, float* __restrict__ lb) {
  __shared__ float tile[16][260];
  int m0 = blockIdx.x * 16;
  int w  = threadIdx.x >> 5;
  int nc0 = (w * 2) * 16, nc1 = nc0 + 16;
  v8f acc0 = v8f_zero(), acc1 = v8f_zero();
  const bf16* A = enc_bf + (size_t)m0 * NH;
  for (int kk = 0; kk < NH; kk += 32) {
    v16bf a = load_a_bf(A, NH, kk);
    acc0 = wmma_bf(a, load_b_bf(Wb1, NH, nc0, kk), acc0);
    acc1 = wmma_bf(a, load_b_bf(Wb1, NH, nc1, kk), acc1);
  }
  int col = threadIdx.x & 15, rb = ((threadIdx.x & 31) >> 4) * 8;
#pragma unroll
  for (int r = 0; r < 8; ++r) {
    int row = rb + r;
    tile[row][nc0 + col] = fmaxf(acc0[r] + b_b1[nc0 + col], 0.f);
    tile[row][nc1 + col] = fmaxf(acc1[r] + b_b1[nc1 + col], 0.f);
  }
  __syncthreads();
  if (threadIdx.x < 16) {
    int row = threadIdx.x;
    float s = 0.f;
    for (int k = 0; k < NH; ++k) s += tile[row][k] * W_b2[k];
    lb[m0 + row] = s + b_b2[0];
  }
}

// per-batch boundary softmax + gumbel sample + cumsum mask bookkeeping
__global__ void k_segB(int seg, const float* __restrict__ lb, const float* __restrict__ gumbel_b,
                       const int* __restrict__ lengths, float* __restrict__ lb0,
                       float* __restrict__ sample_b, float* __restrict__ seg_prob,
                       float* __restrict__ maskws, float* __restrict__ logcum) {
  __shared__ float s[256], red[256];
  int b = blockIdx.x, t = threadIdx.x;
  int len = lengths[b];
  float sb;
  if (seg < NS - 1) {
    float v = lb[b * NT + t];
    if (t == 0 || t >= len) v = C_NEG_INF;
    if (seg == 0) lb0[b * NT + t] = v;
    float e = v + gumbel_b[((size_t)seg * NB + b) * NT + t];   // TEMP_B = 1
    float m = block_max256(e, red);
    float ex = __expf(e - m);
    float den = block_sum256(ex, red);
    sb = ex / den;
  } else {
    sb = (t == len - 1) ? 1.f : 0.f;
  }
  sample_b[b * NT + t] = sb;
  s[t] = sb; __syncthreads();
  for (int o = 1; o < 256; o <<= 1) {                // inclusive scan
    float add = (t >= o) ? s[t - o] : 0.f;
    __syncthreads();
    s[t] += add;
    __syncthreads();
  }
  float cum = s[t];
  float pm = maskws[b * NT + t];                      // masks[seg-1] (ones for seg 0)
  seg_prob[b * NT + t] = (1.f - cum) * pm;
  float lc = logcum[b * NT + t] + logf(cum + C_EPS);
  logcum[b * NT + t] = lc;
  maskws[b * NT + t] = __expf(lc);
}

// readout[b,h] = sum_{t<T-1} enc[b,t,h] * sample_b[b,t+1]
__global__ void k_readout(const bf16* __restrict__ enc_bf, const float* __restrict__ sample_b,
                          float* __restrict__ readout) {
  __shared__ float sb[NT];
  int b = blockIdx.x, h = threadIdx.x;
  sb[h] = sample_b[b * NT + h];
  __syncthreads();
  float acc = 0.f;
  for (int t = 0; t < NT - 1; ++t)
    acc += (float)enc_bf[((size_t)b * NT + t) * NH + h] * sb[t + 1];
  readout[b * NH + h] = acc;
}

// z head: hz=relu(r@Wz1^T+b), lz=hz@Wz2^T+b, KL(z) term, sample_z, Dz = sample_z @ W_d1(z-slice)^T
__global__ void k_z(int seg, const float* __restrict__ readout, const float* __restrict__ Wz1,
                    const float* __restrict__ bz1, const float* __restrict__ Wz2,
                    const float* __restrict__ bz2, const float* __restrict__ gumbel_z,
                    const float* __restrict__ W_d1, float* __restrict__ Dz,
                    float* __restrict__ klz_acc) {
  __shared__ float sr[NH], shh[NH], sz[NH], red[256];
  int b = blockIdx.x, h = threadIdx.x;
  sr[h] = readout[b * NH + h];
  __syncthreads();
  float a = bz1[h];
  for (int k = 0; k < NH; ++k) a += sr[k] * Wz1[h * NH + k];
  shh[h] = fmaxf(a, 0.f);
  __syncthreads();
  float l = bz2[h];
  for (int k = 0; k < NH; ++k) l += shh[k] * Wz2[h * NH + k];
  float m = block_max256(l, red);
  float ex = __expf(l - m);
  float den = block_sum256(ex, red);
  float pz = ex / den;
  float ts = block_sum256(pz * logf(pz + C_EPS), red);
  if (h == 0) atomicAdd(klz_acc, ts / (float)NB);
  float l2 = l + gumbel_z[((size_t)seg * NB + b) * NH + h];   // TEMP_Z = 1
  float m2 = block_max256(l2, red);
  float ex2 = __expf(l2 - m2);
  float den2 = block_sum256(ex2, red);
  sz[h] = ex2 / den2;
  __syncthreads();
  float d = 0.f;
  for (int k = 0; k < NH; ++k) d += sz[k] * W_d1[h * (3 * NH) + 2 * NH + k];
  Dz[b * NH + h] = d;
}

// decoder + NLL: hidden = relu(obs@W_d1o^T + Denv + Dz + b_d1); logits = hidden@W_d2^T + b_d2;
// nll += sum_{t<T-1} (lse - logit[action]) * seg_prob / B
__global__ void k_dec(const bf16* __restrict__ obs_bf, const bf16* __restrict__ Wd1,
                      const float* __restrict__ b_d1, const float* __restrict__ Denv,
                      const float* __restrict__ Dz, const int* __restrict__ env_id,
                      const int* __restrict__ action, const float* __restrict__ W_d2,
                      const float* __restrict__ b_d2, const float* __restrict__ seg_prob,
                      float* __restrict__ nll_acc) {
  __shared__ float tile[16][260];
  __shared__ float lg[16][NA];
  int m0 = blockIdx.x * 16;
  int w  = threadIdx.x >> 5;
  int nc0 = (w * 2) * 16, nc1 = nc0 + 16;
  v8f acc0 = v8f_zero(), acc1 = v8f_zero();
  const bf16* A = obs_bf + (size_t)m0 * NH;
  for (int kk = 0; kk < NH; kk += 32) {
    v16bf a = load_a_bf(A, NH, kk);
    acc0 = wmma_bf(a, load_b_bf(Wd1, 3 * NH, nc0, kk), acc0);
    acc1 = wmma_bf(a, load_b_bf(Wd1, 3 * NH, nc1, kk), acc1);
  }
  int col = threadIdx.x & 15, rb = ((threadIdx.x & 31) >> 4) * 8;
#pragma unroll
  for (int r = 0; r < 8; ++r) {
    int row = rb + r, bt = m0 + row, b = bt >> 8;
    int e = env_id[b];
    float v0 = acc0[r] + b_d1[nc0 + col] + Denv[e * NH + nc0 + col] + Dz[b * NH + nc0 + col];
    float v1 = acc1[r] + b_d1[nc1 + col] + Denv[e * NH + nc1 + col] + Dz[b * NH + nc1 + col];
    tile[row][nc0 + col] = fmaxf(v0, 0.f);
    tile[row][nc1 + col] = fmaxf(v1, 0.f);
  }
  __syncthreads();
  if (threadIdx.x < 16 * NA) {
    int row = threadIdx.x >> 3, a = threadIdx.x & 7;
    float s = b_d2[a];
    for (int k = 0; k < NH; ++k) s += tile[row][k] * W_d2[a * NH + k];
    lg[row][a] = s;
  }
  __syncthreads();
  if (threadIdx.x < 16) {
    int row = threadIdx.x, bt = m0 + row, t = bt & (NT - 1);
    if (t < NT - 1) {
      float mx = lg[row][0];
      for (int a = 1; a < NA; ++a) mx = fmaxf(mx, lg[row][a]);
      float se = 0.f;
      for (int a = 0; a < NA; ++a) se += __expf(lg[row][a] - mx);
      float lse = mx + logf(se);
      float loss = lse - lg[row][action[bt]];
      atomicAdd(nll_acc, loss * seg_prob[bt] / (float)NB);
    }
  }
}

__global__ void k_klb(const float* __restrict__ lb0, const float* __restrict__ logprior,
                      float* __restrict__ klb_acc) {
  __shared__ float red[256];
  int b = blockIdx.x, t = threadIdx.x;
  float v = lb0[b * NT + t];
  float m = block_max256(v, red);
  float ex = __expf(v - m);
  float den = block_sum256(ex, red);
  float p = ex / den;
  float term = (t >= 1) ? p * (logf(p + C_EPS) - logprior[t]) : 0.f;
  float ts = block_sum256(term, red);
  if (t == 0) atomicAdd(klb_acc, (float)NS * ts / (float)NB);
}

__global__ void k_final(const float* __restrict__ accs, float* __restrict__ out) {
  // accs: [0]=nll [1]=kl_z [2]=kl_b ; outputs: loss, nll, kl_b, kl_z
  float nll = accs[0], klz = accs[1], klb = accs[2];
  out[0] = nll + C_BETA_Z * klz + C_BETA_B * klb;
  out[1] = nll;
  out[2] = klb;
  out[3] = klz;
}

// ---------------- host launch ----------------
extern "C" void kernel_launch(void* const* d_in, const int* in_sizes, int n_in,
                              void* d_out, int out_size, void* d_ws, size_t ws_size,
                              hipStream_t stream) {
  (void)in_sizes; (void)n_in; (void)out_size; (void)ws_size;
  const float* features = (const float*)d_in[0];
  const float* gumbel_b = (const float*)d_in[1];
  const float* gumbel_z = (const float*)d_in[2];
  const float* W_obs = (const float*)d_in[3];
  const float* b_obs = (const float*)d_in[4];
  const float* emb_act = (const float*)d_in[5];
  const float* emb_env = (const float*)d_in[6];
  const float* W_enc = (const float*)d_in[7];
  const float* b_enc = (const float*)d_in[8];
  const float* W_ih = (const float*)d_in[9];
  const float* W_hh = (const float*)d_in[10];
  const float* b_ih = (const float*)d_in[11];
  const float* b_hh = (const float*)d_in[12];
  const float* W_z1 = (const float*)d_in[13];
  const float* b_z1 = (const float*)d_in[14];
  const float* W_z2 = (const float*)d_in[15];
  const float* b_z2 = (const float*)d_in[16];
  const float* W_b1 = (const float*)d_in[17];
  const float* b_b1 = (const float*)d_in[18];
  const float* W_b2 = (const float*)d_in[19];
  const float* b_b2 = (const float*)d_in[20];
  const float* W_d1 = (const float*)d_in[21];
  const float* b_d1 = (const float*)d_in[22];
  const float* W_d2 = (const float*)d_in[23];
  const float* b_d2 = (const float*)d_in[24];
  const int* action  = (const int*)d_in[25];
  const int* env_id  = (const int*)d_in[26];
  const int* lengths = (const int*)d_in[27];
  float* out = (float*)d_out;

  char* ws = (char*)d_ws;
  size_t off = 0;
  auto alloc = [&](size_t bytes) -> char* {
    char* p = ws + off;
    off = (off + bytes + 255) & ~(size_t)255;
    return p;
  };
  bf16* obs_bf  = (bf16*)alloc((size_t)NBT * NH * 2);
  bf16* emb_bf  = (bf16*)alloc((size_t)NBT * NH * 2);
  bf16* xg      = (bf16*)alloc((size_t)NBT * NG * 2);
  bf16* enc_bf  = (bf16*)alloc((size_t)NBT * NH * 2);
  bf16* wobs_bf = (bf16*)alloc((size_t)NH * NV * 2);
  bf16* wenc_bf = (bf16*)alloc((size_t)NH * 3 * NH * 2);
  bf16* wih_bf  = (bf16*)alloc((size_t)NG * NH * 2);
  bf16* whh_bf  = (bf16*)alloc((size_t)NG * NH * 2);
  bf16* wb1_bf  = (bf16*)alloc((size_t)NH * NH * 2);
  bf16* wd1_bf  = (bf16*)alloc((size_t)NH * 3 * NH * 2);
  float* E2       = (float*)alloc((size_t)NE * NH * 4);
  float* A3       = (float*)alloc((size_t)NA * NH * 4);
  float* Denv     = (float*)alloc((size_t)NE * NH * 4);
  float* logprior = (float*)alloc((size_t)NT * 4);
  float* lb       = (float*)alloc((size_t)NBT * 4);
  float* lb0      = (float*)alloc((size_t)NBT * 4);
  float* maskws   = (float*)alloc((size_t)NBT * 4);
  float* logcum   = (float*)alloc((size_t)NBT * 4);
  float* sample_b = (float*)alloc((size_t)NBT * 4);
  float* seg_prob = (float*)alloc((size_t)NBT * 4);
  float* readout  = (float*)alloc((size_t)NB * NH * 4);
  float* Dz       = (float*)alloc((size_t)NB * NH * 4);
  float* accs     = (float*)alloc(16 * 4);

  auto cvt = [&](const float* src, bf16* dst, int n) {
    k_f2bf<<<(n + 255) / 256, 256, 0, stream>>>(src, dst, n);
  };
  cvt(W_obs, wobs_bf, NH * NV);
  cvt(W_enc, wenc_bf, NH * 3 * NH);
  cvt(W_ih,  wih_bf,  NG * NH);
  cvt(W_hh,  whh_bf,  NG * NH);
  cvt(W_b1,  wb1_bf,  NH * NH);
  cvt(W_d1,  wd1_bf,  NH * 3 * NH);

  k_init<<<(NBT + 255) / 256, 256, 0, stream>>>(maskws, logcum, accs);
  k_prep<<<1, 256, 0, stream>>>(emb_env, emb_act, W_enc, W_d1, E2, A3, Denv, logprior);

  k_obs<<<NBT / 16, 256, 0, stream>>>(features, wobs_bf, b_obs, obs_bf);
  k_enc<<<NBT / 16, 256, 0, stream>>>(obs_bf, wenc_bf, b_enc, E2, A3, action, env_id, emb_bf);
  k_xproj<<<dim3(NBT / 16, NG / 256), 256, 0, stream>>>(emb_bf, wih_bf, b_ih, b_hh, xg);

  for (int seg = 0; seg < NS; ++seg) {
    k_lstm<<<NB / 16, 512, 0, stream>>>(xg, whh_bf, maskws, enc_bf);
    if (seg < NS - 1)
      k_bound<<<NBT / 16, 256, 0, stream>>>(enc_bf, wb1_bf, b_b1, W_b2, b_b2, lb);
    k_segB<<<NB, 256, 0, stream>>>(seg, lb, gumbel_b, lengths, lb0, sample_b, seg_prob,
                                   maskws, logcum);
    k_readout<<<NB, 256, 0, stream>>>(enc_bf, sample_b, readout);
    k_z<<<NB, 256, 0, stream>>>(seg, readout, W_z1, b_z1, W_z2, b_z2, gumbel_z, W_d1, Dz,
                                accs + 1);
    k_dec<<<NBT / 16, 256, 0, stream>>>(obs_bf, wd1_bf, b_d1, Denv, Dz, env_id, action,
                                        W_d2, b_d2, seg_prob, accs + 0);
  }
  k_klb<<<NB, 256, 0, stream>>>(lb0, logprior, accs + 2);
  k_final<<<1, 1, 0, stream>>>(accs, out);
}
